// attention_net_4045859192980
// MI455X (gfx1250) — compile-verified, matrix operand order checked by reading, not run on hardware
//
#include <hip/hip_runtime.h>
#include <hip/hip_bf16.h>
#include <math.h>

#define BATCH   16
#define TOPN    4
#define OUTS    224
#define PADW    224
#define IMS     448
#define HPS     896
#define NANCH   1614
#define IOU_TH  0.25f

typedef __attribute__((ext_vector_type(16))) _Float16 v16h;
typedef __attribute__((ext_vector_type(8)))  _Float16 v8h;
typedef __attribute__((ext_vector_type(8)))  float    v8f;

union AFrag { v16h v; v8h h2[2]; };
union CFrag { v8f  v; float f[8]; };

__device__ __forceinline__ int imin(int a, int b) { return a < b ? a : b; }

// ---------------------------------------------------------------------------
// Zero-fill an f16 buffer (count must be a multiple of 8 halves / 16 bytes).
// ---------------------------------------------------------------------------
__global__ void zero_f16(_Float16* __restrict__ p, long n16)
{
    const long i = (long)blockIdx.x * blockDim.x + threadIdx.x;
    if (i < n16) ((uint4*)p)[i] = make_uint4(0u, 0u, 0u, 0u);
}

// ---------------------------------------------------------------------------
// Pack NCHW f32 activations into zero-padded channel-last f16:
// out[b][yp][xp][ci], yp/xp in [0, H+2), border = 0.  One thread per element.
// ---------------------------------------------------------------------------
__global__ void pack_act_pad(const float* __restrict__ in, _Float16* __restrict__ outh,
                             int Cin, int Hin, int Win)
{
    const int Hp = Hin + 2, Wp = Win + 2;
    const long total = (long)BATCH * Hp * Wp * Cin;
    const long idx = (long)blockIdx.x * blockDim.x + threadIdx.x;
    if (idx >= total) return;
    int ci   = (int)(idx % Cin);
    long r   = idx / Cin;
    int xp   = (int)(r % Wp); r /= Wp;
    int yp   = (int)(r % Hp);
    int b    = (int)(r / Hp);
    const int y = yp - 1, x = xp - 1;
    _Float16 v = (_Float16)0.0f;
    if (y >= 0 && y < Hin && x >= 0 && x < Win)
        v = (_Float16)in[(((long)b * Cin + ci) * Hin + y) * Win + x];
    outh[idx] = v;
}

// ---------------------------------------------------------------------------
// Pack OIHW f32 3x3 weights into f16 [kyx][co][ci] (B-fragment friendly).
// ---------------------------------------------------------------------------
__global__ void pack_w(const float* __restrict__ w, _Float16* __restrict__ wh,
                       int Cin, int Cout)
{
    const long total = (long)9 * Cout * Cin;
    const long idx = (long)blockIdx.x * blockDim.x + threadIdx.x;
    if (idx >= total) return;
    int ci  = (int)(idx % Cin);
    long r  = idx / Cin;
    int co  = (int)(r % Cout);
    int kyx = (int)(r / Cout);
    wh[idx] = (_Float16)w[((long)co * Cin + ci) * 9 + kyx];
}

// ---------------------------------------------------------------------------
// Implicit-GEMM 3x3 conv from zero-padded channel-last f16 input.
// inh : [B, Hp=Hin+2, Wp=Win+2, Cin] f16 (zero border)
// wh  : [9, Cout, Cin] f16
// out : [B, Cout, Hout, Wout] f32 (bias+ReLU)
// outh: optional [B, Hout+2, Wout+2, Cout] f16 (zero border pre-filled)
// grid.x = (B*Hout*Wout)/16, grid.y = Cout/16, block = 32 (one wave32)
// Inner loop: 2x b128 A-loads + 2x b128 B-loads + 1 WMMA. No branches.
// ---------------------------------------------------------------------------
__global__ void conv3x3_wmma(const _Float16* __restrict__ inh, const _Float16* __restrict__ wh,
                             const float* __restrict__ bias, float* __restrict__ out,
                             _Float16* __restrict__ outh,
                             int Cin, int Hp, int Wp, int Cout, int Hout, int Wout,
                             int stride)
{
    const int l      = threadIdx.x;
    const int m0     = blockIdx.x * 16;
    const int n0     = blockIdx.y * 16;
    const int HWo    = Hout * Wout;
    const int lane16 = l & 15;
    const int kbase  = (l < 16) ? 0 : 8;     // ISA 7.12.2 16-bit A/B lane split

    // A-fragment row for this lane (same row index in both lane halves)
    const int mrow  = m0 + lane16;
    const int b_img = mrow / HWo;
    const int rem   = mrow - b_img * HWo;
    const int oy    = rem / Wout;
    const int ox    = rem - oy * Wout;

    // B-fragment column for this lane
    const int ncol = n0 + lane16;

    CFrag c;
    #pragma unroll
    for (int r = 0; r < 8; ++r) c.f[r] = 0.0f;

    const _Float16* abase = inh + (((long)b_img * Hp + oy * stride) * Wp + ox * stride) * Cin + kbase;
    const _Float16* bbase = wh + (long)ncol * Cin + kbase;

    #pragma unroll
    for (int ky = 0; ky < 3; ++ky) {
        #pragma unroll
        for (int kx = 0; kx < 3; ++kx) {
            const _Float16* ap = abase + ((long)ky * Wp + kx) * Cin;
            const _Float16* bp = bbase + (long)(ky * 3 + kx) * Cout * Cin;
            for (int ci0 = 0; ci0 < Cin; ci0 += 32) {
                AFrag a, bb;
                a.h2[0]  = *(const v8h*)(ap + ci0);        // K = kbase+0..7
                a.h2[1]  = *(const v8h*)(ap + ci0 + 16);   // K = kbase+16..23
                bb.h2[0] = *(const v8h*)(bp + ci0);
                bb.h2[1] = *(const v8h*)(bp + ci0 + 16);
                c.v = __builtin_amdgcn_wmma_f32_16x16x32_f16(
                          false, a.v, false, bb.v, (short)0, c.v, false, false);
            }
        }
    }

    // C/D layout: lanes 0-15 -> M=m0+r, N=lane ; lanes 16-31 -> M=m0+8+r, N=lane-16
    const float bv  = bias[ncol];
    const int   HpO = Hout + 2, WpO = Wout + 2;
    #pragma unroll
    for (int r = 0; r < 8; ++r) {
        const int m   = m0 + r + ((l < 16) ? 0 : 8);
        const int bi  = m / HWo;
        const int sp  = m - bi * HWo;
        float v = c.f[r] + bv;
        v = v > 0.0f ? v : 0.0f;
        out[((long)bi * Cout + ncol) * HWo + sp] = v;
        if (outh) {
            const int oy2 = sp / Wout, ox2 = sp - oy2 * Wout;
            outh[(((long)bi * HpO + oy2 + 1) * WpO + ox2 + 1) * Cout + ncol] = (_Float16)v;
        }
    }
}

// ---------------------------------------------------------------------------
// 1x1 "tidy" convs producing rpn_score [B, 1614] (t1|t2|t3 concat)
// ---------------------------------------------------------------------------
__global__ void tidy_kernel(const float* __restrict__ d1, const float* __restrict__ d2,
                            const float* __restrict__ d3,
                            const float* __restrict__ w1, const float* __restrict__ bb1,
                            const float* __restrict__ w2, const float* __restrict__ bb2,
                            const float* __restrict__ w3, const float* __restrict__ bb3,
                            float* __restrict__ score)
{
    const int idx = blockIdx.x * blockDim.x + threadIdx.x;
    if (idx >= BATCH * NANCH) return;
    const int b = idx / NANCH;
    const int i = idx - b * NANCH;

    const float* d; const float* w; float bias; int strideHW;
    if (i < 1176) {                    // [6,14,14]
        const int cch = i / 196, pp = i - cch * 196;
        d = d1 + (long)b * 128 * 196 + pp; w = w1 + cch * 128; bias = bb1[cch]; strideHW = 196;
    } else if (i < 1470) {             // [6,7,7]
        const int j = i - 1176, cch = j / 49, pp = j - cch * 49;
        d = d2 + (long)b * 128 * 49 + pp;  w = w2 + cch * 128; bias = bb2[cch]; strideHW = 49;
    } else {                           // [9,4,4]
        const int j = i - 1470, cch = j / 16, pp = j - cch * 16;
        d = d3 + (long)b * 128 * 16 + pp;  w = w3 + cch * 128; bias = bb3[cch]; strideHW = 16;
    }
    float s = bias;
    #pragma unroll 4
    for (int ci = 0; ci < 128; ++ci) s += d[ci * strideHW] * w[ci];
    score[idx] = s;
}

// ---------------------------------------------------------------------------
// Greedy hard-NMS: one 256-thread block per image, TOPN sequential rounds.
// ---------------------------------------------------------------------------
__global__ void nms_kernel(const float* __restrict__ score, const int* __restrict__ anchors,
                           float* __restrict__ out_idx_f, float* __restrict__ out_prob,
                           int* __restrict__ ws_idx)
{
    const int b   = blockIdx.x;
    const int tid = threadIdx.x;
    __shared__ float s[NANCH];
    __shared__ float rs[256];
    __shared__ int   ri[256];
    __shared__ float selbox[5];   // y0,x0,y1,x1,area

    for (int i = tid; i < NANCH; i += 256) s[i] = score[b * NANCH + i];
    __syncthreads();

    for (int t = 0; t < TOPN; ++t) {
        float best = -__builtin_inff(); int bi = NANCH;
        for (int i = tid; i < NANCH; i += 256) {
            const float v = s[i];
            if (v > best || (v == best && i < bi)) { best = v; bi = i; }
        }
        rs[tid] = best; ri[tid] = bi;
        __syncthreads();
        for (int off = 128; off > 0; off >>= 1) {
            if (tid < off) {
                const float ov = rs[tid + off]; const int oi = ri[tid + off];
                if (ov > rs[tid] || (ov == rs[tid] && oi < ri[tid])) { rs[tid] = ov; ri[tid] = oi; }
            }
            __syncthreads();
        }
        if (tid == 0) {
            const int si = ri[0];
            const float y0 = (float)anchors[si * 4 + 0], x0 = (float)anchors[si * 4 + 1];
            const float y1 = (float)anchors[si * 4 + 2], x1 = (float)anchors[si * 4 + 3];
            selbox[0] = y0; selbox[1] = x0; selbox[2] = y1; selbox[3] = x1;
            selbox[4] = (y1 - y0) * (x1 - x0);
            out_idx_f[b * TOPN + t] = (float)si;
            out_prob [b * TOPN + t] = rs[0];
            ws_idx   [b * TOPN + t] = si;
        }
        __syncthreads();
        const float sy0 = selbox[0], sx0 = selbox[1], sy1 = selbox[2], sx1 = selbox[3], sa = selbox[4];
        for (int i = tid; i < NANCH; i += 256) {
            const float y0 = (float)anchors[i * 4 + 0], x0 = (float)anchors[i * 4 + 1];
            const float y1 = (float)anchors[i * 4 + 2], x1 = (float)anchors[i * 4 + 3];
            float iy = fminf(y1, sy1) - fmaxf(y0, sy0); iy = iy > 0.0f ? iy : 0.0f;
            float ix = fminf(x1, sx1) - fmaxf(x0, sx0); ix = ix > 0.0f ? ix : 0.0f;
            const float inter = iy * ix;
            const float area  = (y1 - y0) * (x1 - x0);
            const float iou   = inter / (area + sa - inter);
            if (iou >= IOU_TH) s[i] = -__builtin_inff();
        }
        __syncthreads();
    }
}

// ---------------------------------------------------------------------------
// Bilinear crop-resize (align_corners=True) from zero-padded image.
// ---------------------------------------------------------------------------
__global__ void crop_kernel(const float* __restrict__ x, const int* __restrict__ anchors,
                            const int* __restrict__ ws_idx, float* __restrict__ out)
{
    const long p = (long)blockIdx.x * blockDim.x + threadIdx.x;
    const long total = (long)BATCH * TOPN * 3 * OUTS * OUTS;
    if (p >= total) return;

    const int crop = (int)(p / (3 * OUTS * OUTS));
    const int rem  = (int)(p - (long)crop * (3 * OUTS * OUTS));
    const int cch  = rem / (OUTS * OUTS);
    const int r2   = rem - cch * (OUTS * OUTS);
    const int oy   = r2 / OUTS;
    const int ox   = r2 - oy * OUTS;
    const int b    = crop / TOPN;

    const int sel = ws_idx[crop];
    const float y0 = (float)anchors[sel * 4 + 0], x0 = (float)anchors[sel * 4 + 1];
    const float y1 = (float)anchors[sel * 4 + 2], x1 = (float)anchors[sel * 4 + 3];

    const float inv = 1.0f / (float)(OUTS - 1);
    const float ys = y0 + ((float)oy * inv) * (y1 - y0 - 1.0f);
    const float xs = x0 + ((float)ox * inv) * (x1 - x0 - 1.0f);

    const int yi0 = (int)floorf(ys); const int yi1 = imin(yi0 + 1, HPS - 1);
    const int xi0 = (int)floorf(xs); const int xi1 = imin(xi0 + 1, HPS - 1);
    const float wy = ys - (float)yi0;
    const float wx = xs - (float)xi0;

    const float* xc = x + ((long)b * 3 + cch) * IMS * IMS;
    auto g = [&](int yi, int xi) -> float {
        const int yy = yi - PADW, xx = xi - PADW;
        if (yy < 0 || yy >= IMS || xx < 0 || xx >= IMS) return 0.0f;
        return xc[yy * IMS + xx];
    };

    const float v = g(yi0, xi0) * (1.0f - wy) * (1.0f - wx)
                  + g(yi0, xi1) * (1.0f - wy) * wx
                  + g(yi1, xi0) * wy * (1.0f - wx)
                  + g(yi1, xi1) * wy * wx;
    out[p] = v;
}

// ---------------------------------------------------------------------------
extern "C" void kernel_launch(void* const* d_in, const int* in_sizes, int n_in,
                              void* d_out, int out_size, void* d_ws, size_t ws_size,
                              hipStream_t stream)
{
    const float* x    = (const float*)d_in[0];
    const float* rpn  = (const float*)d_in[1];
    const float* w1   = (const float*)d_in[2];  const float* b1  = (const float*)d_in[3];
    const float* w2   = (const float*)d_in[4];  const float* b2  = (const float*)d_in[5];
    const float* w3   = (const float*)d_in[6];  const float* b3  = (const float*)d_in[7];
    const float* wt1  = (const float*)d_in[8];  const float* bt1 = (const float*)d_in[9];
    const float* wt2  = (const float*)d_in[10]; const float* bt2 = (const float*)d_in[11];
    const float* wt3  = (const float*)d_in[12]; const float* bt3 = (const float*)d_in[13];
    const int*   anc  = (const int*)d_in[14];

    float* out = (float*)d_out;
    const long OFF_IDX   = (long)BATCH * TOPN * 3 * OUTS * OUTS;   // 9,633,792
    const long OFF_PROB  = OFF_IDX  + BATCH * TOPN;
    const long OFF_SCORE = OFF_PROB + BATCH * TOPN;

    // ---- workspace layout (16B-aligned sections) ----
    float* ws   = (float*)d_ws;
    float* d1   = ws;                                   // [16,128,14,14] f32
    float* d2   = d1 + (long)BATCH * 128 * 196;         // [16,128,7,7]  f32
    float* d3   = d2 + (long)BATCH * 128 * 49;          // [16,128,4,4]  f32
    int*   wsidx = (int*)(d3 + (long)BATCH * 128 * 16); // 64 ints (pad to 64 floats)
    _Float16* hbase = (_Float16*)((float*)wsidx + 64);
    const long N_AHP1 = (long)BATCH * 16 * 16 * 2048;   // padded rpn, channel-last f16
    const long N_D1H  = (long)BATCH * 16 * 16 * 128;    // padded d1 f16
    const long N_D2H  = (long)BATCH * 9  * 9  * 128;    // padded d2 f16
    const long N_WH1  = (long)9 * 128 * 2048;
    const long N_WH2  = (long)9 * 128 * 128;
    const long N_WH3  = (long)9 * 128 * 128;
    _Float16* ahp1 = hbase;
    _Float16* d1h  = ahp1 + N_AHP1;
    _Float16* d2h  = d1h  + N_D1H;
    _Float16* wh1  = d2h  + N_D2H;
    _Float16* wh2  = wh1  + N_WH1;
    _Float16* wh3  = wh2  + N_WH2;

    // ---- prep: zero padded f16 intermediates (borders), pack act + weights ----
    zero_f16<<<(int)((N_D1H / 8 + 255) / 256), 256, 0, stream>>>(d1h, N_D1H / 8);
    zero_f16<<<(int)((N_D2H / 8 + 255) / 256), 256, 0, stream>>>(d2h, N_D2H / 8);
    pack_act_pad<<<(int)((N_AHP1 + 255) / 256), 256, 0, stream>>>(rpn, ahp1, 2048, 14, 14);
    pack_w<<<(int)((N_WH1 + 255) / 256), 256, 0, stream>>>(w1, wh1, 2048, 128);
    pack_w<<<(int)((N_WH2 + 255) / 256), 256, 0, stream>>>(w2, wh2, 128, 128);
    pack_w<<<(int)((N_WH3 + 255) / 256), 256, 0, stream>>>(w3, wh3, 128, 128);

    // ---- conv tower (WMMA implicit GEMM) ----
    // conv1: 2048->128, 14x14, stride 1 (M=3136 -> 196 tiles)
    conv3x3_wmma<<<dim3(196, 8), 32, 0, stream>>>(ahp1, wh1, b1, d1, d1h,
                                                  2048, 16, 16, 128, 14, 14, 1);
    // conv2: 128->128, 14->7, stride 2 (M=784 -> 49 tiles)
    conv3x3_wmma<<<dim3(49, 8), 32, 0, stream>>>(d1h, wh2, b2, d2, d2h,
                                                 128, 16, 16, 128, 7, 7, 2);
    // conv3: 128->128, 7->4, stride 2 (M=256 -> 16 tiles)
    conv3x3_wmma<<<dim3(16, 8), 32, 0, stream>>>(d2h, wh3, b3, d3, (_Float16*)nullptr,
                                                 128, 9, 9, 128, 4, 4, 2);

    // ---- heads ----
    tidy_kernel<<<(BATCH * NANCH + 255) / 256, 256, 0, stream>>>(
        d1, d2, d3, wt1, bt1, wt2, bt2, wt3, bt3, out + OFF_SCORE);

    nms_kernel<<<BATCH, 256, 0, stream>>>(out + OFF_SCORE, anc,
                                          out + OFF_IDX, out + OFF_PROB, wsidx);

    const long total = (long)BATCH * TOPN * 3 * OUTS * OUTS;
    crop_kernel<<<(int)((total + 255) / 256), 256, 0, stream>>>(x, anc, wsidx, out);
}
